// VariationalDecoder_meta_predvar_1151051235838
// MI455X (gfx1250) — compile-verified, matrix operand order checked by reading, not run on hardware
//
#include <hip/hip_runtime.h>
#include <hip/hip_bf16.h>

// ---------------------------------------------------------------------------
// Problem constants (from reference)
// ---------------------------------------------------------------------------
#define BATCH 2048
#define LSEQ  168
#define HDIM  256     // hidden
#define XPDIM 128     // x_prime
#define DXL   32
#define DXT   16
#define DXW   16
#define DXS   16
#define XCAT  80      // DXL+DXT+DXW+DXS
#define XCATP 96      // padded to multiple of 32
#define KMETA 336     // HDIM + XCAT
#define KMETAP 352    // padded
#define G3    768     // 3*HDIM
#define OUTN  24
#define OUTNP 32
#define TSTEPS (LSEQ + 1)   // 169 output time points
#define OUT_ROW_STRIDE ((long)TSTEPS * OUTN)

#define NTHREADS 128        // 4 waves per workgroup

// ---------------------------------------------------------------------------
// WMMA types
// ---------------------------------------------------------------------------
typedef __attribute__((ext_vector_type(16))) __bf16 v16bf;
typedef __attribute__((ext_vector_type(8)))  float  v8f;
typedef __attribute__((ext_vector_type(4)))  float  v4f;

union FragU {
    v16bf v;
    uint4 u[2];
};

__device__ __forceinline__ v8f wmma_bf16(v16bf a, v16bf b, v8f c) {
    return __builtin_amdgcn_wmma_f32_16x16x32_bf16(
        /*neg_a=*/false, a, /*neg_b=*/false, b,
        /*c_mod=*/(short)0, c, /*reuse_a=*/false, /*reuse_b=*/false);
}

__device__ __forceinline__ unsigned short f2bf(float f) {
    unsigned int x = __float_as_uint(f);
    unsigned int lsb = (x >> 16) & 1u;
    return (unsigned short)((x + 0x7FFFu + lsb) >> 16);
}

__device__ __forceinline__ float sigm(float x) {
    return 1.0f / (1.0f + __expf(-x));
}

// A-fragment (16x32 bf16, M=lane%16, K split by lane-half) from row-major
// bf16(ushort) LDS buffer; stride in elements, k0 multiple of 32.
__device__ __forceinline__ v16bf lds_fragA(const unsigned short* buf, int stride, int k0) {
    int l = threadIdx.x & 31;
    const unsigned short* p = buf + (l & 15) * stride + k0 + ((l >> 4) << 3);
    FragU f;
    f.u[0] = *(const uint4*)(p);
    f.u[1] = *(const uint4*)(p + 16);
    return f.v;
}

// B-fragment (32x16 bf16, N=lane%16, K split by lane-half) from row-major
// weight table W[Nout][Kpad] in global memory (k-major per output row).
__device__ __forceinline__ v16bf gmem_fragB(const unsigned short* W, int stride, int n0, int k0) {
    int l = threadIdx.x & 31;
    const unsigned short* p = W + (long)(n0 + (l & 15)) * stride + k0 + ((l >> 4) << 3);
    FragU f;
    f.u[0] = *(const uint4*)(p);
    f.u[1] = *(const uint4*)(p + 16);
    return f.v;
}

// ---------------------------------------------------------------------------
// Weight conversion: f32 [srows][scols] -> bf16 [drows][dcols], zero padded
// ---------------------------------------------------------------------------
__global__ void cvt_pad_bf16(const float* __restrict__ src, unsigned short* __restrict__ dst,
                             int drows, int srows, int scols, int dcols) {
    int i = blockIdx.x * blockDim.x + threadIdx.x;
    if (i >= drows * dcols) return;
    int r = i / dcols, c = i % dcols;
    float v = (r < srows && c < scols) ? src[(long)r * scols + c] : 0.0f;
    dst[i] = f2bf(v);
}

// ---------------------------------------------------------------------------
// One head tile: 16 output columns of (h @ W.T + b) for one head.
// D-fragment layout: lane -> output col (within 16-tile), vgpr r -> batch row
// ---------------------------------------------------------------------------
__device__ __forceinline__ void do_head_tile(const v16bf* hA, const unsigned short* Wb,
                                             const float* bias, float* outp,
                                             int nt, int t, int b0) {
    int l = threadIdx.x & 31;
    int col_l = l & 15;
    int rhalf = (l >> 4) << 3;
    int col = nt * 16 + col_l;
    float bv = (col < OUTN) ? bias[col] : 0.0f;
    v8f c = {bv, bv, bv, bv, bv, bv, bv, bv};
#pragma unroll
    for (int k = 0; k < 8; ++k)
        c = wmma_bf16(hA[k], gmem_fragB(Wb, HDIM, nt * 16, k * 32), c);
    if (col < OUTN) {
#pragma unroll
        for (int r = 0; r < 8; ++r) {
            int row = r + rhalf;
            outp[(long)(b0 + row) * OUT_ROW_STRIDE + (long)t * OUTN + col] = c[r];
        }
    }
}

// ---------------------------------------------------------------------------
// Main GRU kernel: one workgroup (4 waves) owns one 16-row batch tile for
// all L timesteps.  The feature dimension of every stage is split across
// the 4 waves; LDS holds the shared hidden state and activations.
// ---------------------------------------------------------------------------
__global__ void __launch_bounds__(NTHREADS)
gru_decoder_kernel(const float* __restrict__ xl, const float* __restrict__ xt,
                   const float* __restrict__ xw, const float* __restrict__ xs,
                   const float* __restrict__ z,
                   const float* __restrict__ b_meta, const float* __restrict__ b_ih,
                   const float* __restrict__ b_hh, const float* __restrict__ b_mu,
                   const float* __restrict__ b_lv,
                   const unsigned short* __restrict__ WmetaB,
                   const unsigned short* __restrict__ WihB,
                   const unsigned short* __restrict__ WhhB,
                   const unsigned short* __restrict__ WmuB,
                   const unsigned short* __restrict__ WlvB,
                   float* __restrict__ out_mu, float* __restrict__ out_lv) {
    __shared__ float          h32[16 * HDIM];    // fp32 master hidden state
    __shared__ unsigned short hbf[16 * HDIM];    // bf16 copy for WMMA A-frags
    __shared__ unsigned short xbuf[16 * XCATP];  // per-step x inputs (bf16, padded)
    __shared__ unsigned short xpbuf[16 * XPDIM]; // x_prime (bf16)

    const int tid   = threadIdx.x;       // 0..127
    const int lane  = tid & 31;
    const int wv    = tid >> 5;          // wave id 0..3
    const int col_l = lane & 15;
    const int rhalf = (lane >> 4) << 3;
    const int b0    = blockIdx.x * 16;

    // Warm L2 with recurrent weights (global_prefetch_b8)
    for (int off = tid * 64; off < G3 * HDIM; off += NTHREADS * 64)
        __builtin_prefetch(WhhB + off, 0, 1);
    for (int off = tid * 64; off < G3 * XPDIM; off += NTHREADS * 64)
        __builtin_prefetch(WihB + off, 0, 1);

    // Stage z_latent -> h (fp32 + bf16): 1024 float4 over 128 threads
    for (int it = 0; it < 8; ++it) {
        int idx4 = tid + NTHREADS * it;
        int row  = idx4 >> 6;
        int c4   = (idx4 & 63) * 4;
        v4f v = *(const v4f*)(z + (long)(b0 + row) * HDIM + c4);
        *(v4f*)(h32 + row * HDIM + c4) = v;
        hbf[row * HDIM + c4 + 0] = f2bf(v[0]);
        hbf[row * HDIM + c4 + 1] = f2bf(v[1]);
        hbf[row * HDIM + c4 + 2] = f2bf(v[2]);
        hbf[row * HDIM + c4 + 3] = f2bf(v[3]);
    }
    // Zero xbuf padding columns [80, 96): 256 elems over 128 threads
    for (int it = 0; it < 2; ++it) {
        int i = tid + NTHREADS * it;
        xbuf[(i >> 4) * XCATP + 80 + (i & 15)] = 0;
    }
    __syncthreads();

    // Per-wave head job: wave 0: mu tile0, 1: mu tile1, 2: lv tile0, 3: lv tile1
    const unsigned short* Whead = (wv & 2) ? WlvB : WmuB;
    const float*          bhead = (wv & 2) ? b_lv : b_mu;
    float*                ohead = (wv & 2) ? out_lv : out_mu;
    const int             nthead = wv & 1;

    v16bf hA[8];
#pragma unroll
    for (int k = 0; k < 8; ++k) hA[k] = lds_fragA(hbf, HDIM, k * 32);

    // t = 0 heads from z_latent
    do_head_tile(hA, Whead, bhead, ohead, nthead, 0, b0);

    for (int t = 0; t < LSEQ; ++t) {
        // ---- stage x(t) into xbuf (bf16): 320 float4 over 128 threads ----
        for (int it = 0; it < 3; ++it) {
            int i = tid + NTHREADS * it;
            if (i < 320) {
                int row = i / 20;
                int c4  = i % 20;
                const float* src;
                int coloff;
                long base = (long)(b0 + row) * LSEQ + t;
                if (c4 < 8)       { src = xl + base * DXL + (c4 - 0) * 4;  coloff = c4 * 4; }
                else if (c4 < 12) { src = xt + base * DXT + (c4 - 8) * 4;  coloff = 32 + (c4 - 8) * 4; }
                else if (c4 < 16) { src = xw + base * DXW + (c4 - 12) * 4; coloff = 48 + (c4 - 12) * 4; }
                else              { src = xs + base * DXS + (c4 - 16) * 4; coloff = 64 + (c4 - 16) * 4; }
                v4f v = *(const v4f*)src;
                unsigned short* d = xbuf + row * XCATP + coloff;
                d[0] = f2bf(v[0]); d[1] = f2bf(v[1]); d[2] = f2bf(v[2]); d[3] = f2bf(v[3]);
            }
        }
        __syncthreads();   // barrier A: x staged

        // ---- meta: x' = tanh([h, x] @ W_meta.T + b_meta), 2 tiles per wave ----
        v16bf xA[3];
#pragma unroll
        for (int k = 0; k < 3; ++k) xA[k] = lds_fragA(xbuf, XCATP, k * 32);

#pragma unroll
        for (int nn_ = 0; nn_ < 2; ++nn_) {
            int n = wv * 2 + nn_;
            float bv = b_meta[n * 16 + col_l];
            v8f c = {bv, bv, bv, bv, bv, bv, bv, bv};
#pragma unroll
            for (int k = 0; k < 8; ++k)
                c = wmma_bf16(hA[k], gmem_fragB(WmetaB, KMETAP, n * 16, k * 32), c);
#pragma unroll
            for (int k = 0; k < 3; ++k)
                c = wmma_bf16(xA[k], gmem_fragB(WmetaB, KMETAP, n * 16, HDIM + k * 32), c);
#pragma unroll
            for (int r = 0; r < 8; ++r)
                xpbuf[(r + rhalf) * XPDIM + n * 16 + col_l] = f2bf(tanhf(c[r]));
        }
        __syncthreads();   // barrier B: x' complete

        v16bf pA[4];
#pragma unroll
        for (int k = 0; k < 4; ++k) pA[k] = lds_fragA(xpbuf, XPDIM, k * 32);

        // ---- GRU gates: 4 hidden-feature tiles j per wave ----
        for (int jj = 0; jj < 4; ++jj) {
            const int j  = wv * 4 + jj;
            const int nr = 0 * HDIM + j * 16;
            const int nz = 1 * HDIM + j * 16;
            const int nn = 2 * HDIM + j * 16;
            float bxr = b_ih[nr + col_l], bxz = b_ih[nz + col_l], bxn = b_ih[nn + col_l];
            float bhr = b_hh[nr + col_l], bhz = b_hh[nz + col_l], bhn = b_hh[nn + col_l];
            v8f gxr = {bxr, bxr, bxr, bxr, bxr, bxr, bxr, bxr};
            v8f gxz = {bxz, bxz, bxz, bxz, bxz, bxz, bxz, bxz};
            v8f gxn = {bxn, bxn, bxn, bxn, bxn, bxn, bxn, bxn};
            v8f ghr = {bhr, bhr, bhr, bhr, bhr, bhr, bhr, bhr};
            v8f ghz = {bhz, bhz, bhz, bhz, bhz, bhz, bhz, bhz};
            v8f ghn = {bhn, bhn, bhn, bhn, bhn, bhn, bhn, bhn};
#pragma unroll
            for (int k = 0; k < 4; ++k) {
                gxr = wmma_bf16(pA[k], gmem_fragB(WihB, XPDIM, nr, k * 32), gxr);
                gxz = wmma_bf16(pA[k], gmem_fragB(WihB, XPDIM, nz, k * 32), gxz);
                gxn = wmma_bf16(pA[k], gmem_fragB(WihB, XPDIM, nn, k * 32), gxn);
            }
#pragma unroll
            for (int k = 0; k < 8; ++k) {
                ghr = wmma_bf16(hA[k], gmem_fragB(WhhB, HDIM, nr, k * 32), ghr);
                ghz = wmma_bf16(hA[k], gmem_fragB(WhhB, HDIM, nz, k * 32), ghz);
                ghn = wmma_bf16(hA[k], gmem_fragB(WhhB, HDIM, nn, k * 32), ghn);
            }
#pragma unroll
            for (int r = 0; r < 8; ++r) {
                int row  = r + rhalf;
                int hidx = row * HDIM + j * 16 + col_l;
                float hold = h32[hidx];
                float rg = sigm(gxr[r] + ghr[r]);
                float zg = sigm(gxz[r] + ghz[r]);
                float ng = tanhf(gxn[r] + rg * ghn[r]);
                float hn = (1.0f - zg) * ng + zg * hold;
                h32[hidx] = hn;
                hbf[hidx] = f2bf(hn);
            }
        }
        __syncthreads();   // barrier C: h(t+1) complete

        // ---- refresh hA from new hidden state, emit this wave's head tile ----
#pragma unroll
        for (int k = 0; k < 8; ++k) hA[k] = lds_fragA(hbf, HDIM, k * 32);
        do_head_tile(hA, Whead, bhead, ohead, nthead, t + 1, b0);
    }
}

// ---------------------------------------------------------------------------
// Launch
// ---------------------------------------------------------------------------
extern "C" void kernel_launch(void* const* d_in, const int* in_sizes, int n_in,
                              void* d_out, int out_size, void* d_ws, size_t ws_size,
                              hipStream_t stream) {
    const float* xl     = (const float*)d_in[0];
    const float* xt     = (const float*)d_in[1];
    const float* xw     = (const float*)d_in[2];
    const float* xs     = (const float*)d_in[3];
    const float* z      = (const float*)d_in[4];
    const float* W_meta = (const float*)d_in[5];
    const float* b_meta = (const float*)d_in[6];
    const float* W_ih   = (const float*)d_in[7];
    const float* W_hh   = (const float*)d_in[8];
    const float* b_ih   = (const float*)d_in[9];
    const float* b_hh   = (const float*)d_in[10];
    const float* W_mu   = (const float*)d_in[11];
    const float* b_mu   = (const float*)d_in[12];
    const float* W_lv   = (const float*)d_in[13];
    const float* b_lv   = (const float*)d_in[14];

    // bf16 weight tables in workspace (16B-aligned offsets)
    char* ws = (char*)d_ws;
    unsigned short* WmetaB = (unsigned short*)(ws + 0);        // 128 x 352  ->  90112 B
    unsigned short* WihB   = (unsigned short*)(ws + 90112);    // 768 x 128  -> 196608 B
    unsigned short* WhhB   = (unsigned short*)(ws + 286720);   // 768 x 256  -> 393216 B
    unsigned short* WmuB   = (unsigned short*)(ws + 679936);   //  32 x 256  ->  16384 B
    unsigned short* WlvB   = (unsigned short*)(ws + 696320);   //  32 x 256  ->  16384 B

    auto cvt = [&](const float* src, unsigned short* dst, int drows, int srows,
                   int scols, int dcols) {
        int total = drows * dcols;
        cvt_pad_bf16<<<(total + 255) / 256, 256, 0, stream>>>(src, dst, drows, srows,
                                                              scols, dcols);
    };
    cvt(W_meta, WmetaB, XPDIM, XPDIM, KMETA, KMETAP);
    cvt(W_ih,   WihB,   G3,    G3,    XPDIM, XPDIM);
    cvt(W_hh,   WhhB,   G3,    G3,    HDIM,  HDIM);
    cvt(W_mu,   WmuB,   OUTNP, OUTN,  HDIM,  HDIM);
    cvt(W_lv,   WlvB,   OUTNP, OUTN,  HDIM,  HDIM);

    float* out_mu = (float*)d_out;
    float* out_lv = out_mu + (long)BATCH * TSTEPS * OUTN;

    gru_decoder_kernel<<<BATCH / 16, NTHREADS, 0, stream>>>(
        xl, xt, xw, xs, z, b_meta, b_ih, b_hh, b_mu, b_lv,
        WmetaB, WihB, WhhB, WmuB, WlvB, out_mu, out_lv);
}
